// RowMAB_14843406975012
// MI455X (gfx1250) — matrix operand order repaired and checked by reading, not compiled, not run on hardware
//
#include <hip/hip_runtime.h>
#include <hip/hip_fp16.h>

// MI455X (gfx1250) wave32 WMMA implementation of RowMAB.
// Workspace layout (requires ws_size >= 38 MB):
//   [ 0,16MB) qh    f16 [n][h][i][kappa]   kappa = r*16+d, 32*2048*128
//   [16,32MB) kh    f16 [n][h][j][kappa]
//   [32,34MB) vbarT f16 [n][h][d][j]       (transposed for PV B-fragments)
//   [34,36MB) ksumh f16 [j*4+n][e]         (K pre-reduced over r)
//   [36,38MB) ctxh  f16 [i*4+n][e]

typedef _Float16 f16;
typedef __attribute__((ext_vector_type(16))) _Float16 v16h;
typedef __attribute__((ext_vector_type(8)))  _Float16 v8h;
typedef __attribute__((ext_vector_type(8)))  float    v8f;
typedef __attribute__((ext_vector_type(4)))  unsigned int v4u;
typedef __attribute__((ext_vector_type(8)))  int      v8i;
typedef __attribute__((ext_vector_type(4)))  int      v4i;

#define WMMA16(a, b, c) \
  __builtin_amdgcn_wmma_f32_16x16x32_f16(false, (a), false, (b), (short)0, (c), false, false)

__device__ __forceinline__ v16h cat8(v8h a, v8h b) {
  v16h r;
#pragma unroll
  for (int u = 0; u < 8; ++u) { r[u] = a[u]; r[u + 8] = b[u]; }
  return r;
}

// ---------------------------------------------------------------------------
// Tensor Data Mover: DMA one contiguous 32 KB tile (8192 x 4B) global -> LDS.
// D# group0: count=1, lds_addr, 57-bit global addr, type=2 ("image").
// D# group1: data_size=4B, tensor_dim0=tile_dim0=8192, tensor_dim1=tile_dim1=1,
//            tensor_dim0_stride=8192, no multicast / barrier / pad / iterate.
// Groups 2/3 zero (<=2D tensor).  Tracked with TENSORcnt.
// This toolchain exposes the 6-arg builtin: (g0, g1, g2, g3, ext, cpol).
// ---------------------------------------------------------------------------
__device__ __forceinline__ void tdm_load_tile32k(unsigned int ldsOff, const void* gptr) {
  unsigned long long ga = (unsigned long long)(uintptr_t)gptr;
  v4u g0 = { 1u,                                   // count=1 (valid), user mode
             ldsOff,                               // lds_addr[31:0]
             (unsigned int)ga,                     // global_addr[31:0]
             (unsigned int)((ga >> 32) & 0x1FFFFFFu) | (2u << 30) }; // addr[56:32] | type=2
  v8i g1 = { (int)(2u << 16),                      // wg_mask=0, data_size=2 (4B)
             (int)(8192u << 16),                   // tensor_dim0[15:0]  (bits 63:48)
             (int)(1u << 16),                      // tensor_dim0[31:16]=0 | tensor_dim1=1
             (int)(8192u << 16),                   // tensor_dim1 hi=0   | tile_dim0=8192
             1,                                    // tile_dim1=1 | tile_dim2=0
             8192,                                 // tensor_dim0_stride[31:0]
             0, 0 };                               // stride0 hi / tensor_dim1_stride
  v4i z4 = { 0, 0, 0, 0 };
  v8i z8 = { 0, 0, 0, 0, 0, 0, 0, 0 };
  __builtin_amdgcn_tensor_load_to_lds(g0, g1, z4, z4, z8, 0);
}

// ---------------------------------------------------------------------------
// Pre-reduce K over r:  ksumh[s][e] = sum_r K[(r*8192 + s)*128 + e], s=j*4+n
// ---------------------------------------------------------------------------
__global__ __launch_bounds__(256) void ksum_kernel(const float* __restrict__ K,
                                                   f16* __restrict__ ksumh) {
  int idx = blockIdx.x * 256 + threadIdx.x;   // over 8192*128
  int e = idx & 127, s = idx >> 7;
  float acc = 0.f;
#pragma unroll
  for (int r = 0; r < 8; ++r) acc += K[(size_t)(r * 8192 + s) * 128 + e];
  ksumh[idx] = (f16)acc;
}

// ---------------------------------------------------------------------------
// WMMA projection: out_rows = A @ W^T + bias.  One wave -> 16 rows x 128 cols.
// MODE 0: scatter f16 into qh/kh layout [n][h][i][kappa]
// MODE 1: scatter f16 into vbarT layout [n][h][d][j]
// MODE 2: f32 rows straight to d_out
// ---------------------------------------------------------------------------
template <int MODE>
__global__ __launch_bounds__(256)
void proj_kernel(const float* __restrict__ Af32, const f16* __restrict__ Af16,
                 const float* __restrict__ W, const float* __restrict__ bias,
                 float bscale, f16* __restrict__ dst16, float* __restrict__ dst32) {
  __shared__ f16  sW[128 * 128];   // 32 KB, W row-major [n][k] as f16
  __shared__ float sB[128];

  const int tid = threadIdx.x;
  for (int q = tid; q < 128 * 128; q += 256) sW[q] = (f16)W[q];
  if (tid < 128) sB[tid] = bias[tid] * bscale;
  __syncthreads();

  const int wave = tid >> 5, lane = tid & 31;
  const int l15 = lane & 15, half = lane >> 4;
  const int r0 = blockIdx.x * 128 + wave * 16;
  const int arow = r0 + l15;

  // A fragments: lane holds row (l15), K chunk {c0..c0+15}
  v16h af[4];
#pragma unroll
  for (int t = 0; t < 4; ++t) {
    const int c0 = t * 32 + half * 16;
    if (MODE == 0) {                                  // f32 source (Q or K)
      const float* p = Af32 + (size_t)arow * 128 + c0;
      v8f x0 = *(const v8f*)p;
      v8f x1 = *(const v8f*)(p + 8);
#pragma unroll
      for (int u = 0; u < 8; ++u) { af[t][u] = (f16)x0[u]; af[t][u + 8] = (f16)x1[u]; }
    } else {                                          // f16 source (ksumh / ctxh)
      const f16* p = Af16 + (size_t)arow * 128 + c0;
      af[t] = cat8(*(const v8h*)p, *(const v8h*)(p + 8));
    }
  }

#pragma unroll
  for (int nb = 0; nb < 8; ++nb) {
    v8f acc = {};
#pragma unroll
    for (int t = 0; t < 4; ++t) {
      // B fragment: lane = column (nb*16+l15), 16 contiguous K values
      const f16* p = sW + (size_t)(nb * 16 + l15) * 128 + t * 32 + half * 16;
      acc = WMMA16(af[t], *(const v16h*)p, acc);
    }
    const int   col = nb * 16 + l15;
    const float bv  = sB[col];
#pragma unroll
    for (int v = 0; v < 8; ++v) {
      const float val = acc[v] + bv;
      const int rowAbs = r0 + v + half * 8;           // C-layout: lanes>=16 -> M+8
      if (MODE == 0) {
        const int n = rowAbs & 3, i = (rowAbs >> 2) & 2047, rr = rowAbs >> 13;
        const int h = col >> 4, d = col & 15;
        dst16[((size_t)(n * 8 + h) * 2048 + i) * 128 + rr * 16 + d] = (f16)val;
      } else if (MODE == 1) {
        const int n = rowAbs & 3, j = rowAbs >> 2;
        const int h = col >> 4, d = col & 15;
        dst16[((size_t)(n * 8 + h) * 16 + d) * 2048 + j] = (f16)val;
      } else {
        dst32[(size_t)rowAbs * 128 + col] = val;
      }
    }
  }
}

// ---------------------------------------------------------------------------
// Fused flash attention per (n,h): S = Q K^T (K=128), online softmax, P*Vbar.
// Block = 8 waves x 16 i-rows = 128-row i-tile; loop over 16 j-tiles of 128.
// K-tiles are double-buffered in LDS and moved by the Tensor Data Mover
// (tensor_load_to_lds, TENSORcnt), overlapping DMA of tile jt+1 with WMMA on jt.
// Dynamic LDS: [0,32K) K buf0 | [32K,64K) K buf1 | [64K,96K) per-wave P tiles.
// NOTE: buffer pointers are derived from `smem` at each use (no initialized
// pointer array -> no addrspacecast in a static initializer).
// ---------------------------------------------------------------------------
__global__ __launch_bounds__(256)
void attn_kernel(const f16* __restrict__ qh, const f16* __restrict__ kh,
                 const f16* __restrict__ vbarT, f16* __restrict__ ctxh) {
  extern __shared__ char smem[];

  const int tid = threadIdx.x, wave = tid >> 5, lane = tid & 31;
  const int l15 = lane & 15, half = lane >> 4;
  const int nh = blockIdx.y;                    // n = nh>>3, h = nh&7
  const int n = nh >> 3, h = nh & 7;
  const int i0 = blockIdx.x * 128 + wave * 16;

  f16* pw = (f16*)(smem + 65536) + (size_t)wave * (16 * 128);

  const f16* qbase = qh + (size_t)nh * 2048 * 128;
  const f16* kbase = kh + (size_t)nh * 2048 * 128;
  const f16* vbase = vbarT + (size_t)nh * 16 * 2048;

  // Q A-fragments for this wave's 16 rows, reused across all j-tiles.
  v16h af[4];
  {
    const int arow = i0 + l15;
#pragma unroll
    for (int t = 0; t < 4; ++t) {
      const f16* p = qbase + (size_t)arow * 128 + t * 32 + half * 16;
      af[t] = cat8(*(const v8h*)p, *(const v8h*)(p + 8));
    }
  }

  v8f   ctx = {};
  float m[8], lsum[8];
#pragma unroll
  for (int v = 0; v < 8; ++v) { m[v] = -3.0e38f; lsum[v] = 0.f; }

  // Prime the pipeline: DMA j-tile 0 into buffer 0 (single wave issues; TDM
  // ignores EXEC, completion tracked with TENSORcnt).
  if (wave == 0)
    tdm_load_tile32k((unsigned int)(uintptr_t)(void*)smem, kbase);

  for (int jt = 0; jt < 16; ++jt) {
    __syncthreads();   // all waves done reading the buffer we are about to refill
    if (wave == 0) {
      if (jt + 1 < 16) {
        tdm_load_tile32k(
            (unsigned int)(uintptr_t)(void*)(smem + (size_t)((jt + 1) & 1) * 32768),
            kbase + (size_t)(jt + 1) * 128 * 128);
        __builtin_amdgcn_s_wait_tensorcnt(1);   // tile jt landed; jt+1 in flight
      } else {
        __builtin_amdgcn_s_wait_tensorcnt(0);
      }
    }
    __syncthreads();   // tile jt visible to every wave
    const f16* sK = (const f16*)(smem + (size_t)(jt & 1) * 32768);

    // S tile: 16 x 128 in eight 16x16 f32 accumulators
    v8f s[8];
#pragma unroll
    for (int jb = 0; jb < 8; ++jb) {
      s[jb] = (v8f){};
#pragma unroll
      for (int t = 0; t < 4; ++t) {
        const f16* p = sK + (size_t)(jb * 16 + l15) * 128 + t * 32 + half * 16;
        s[jb] = WMMA16(af[t], *(const v16h*)p, s[jb]);
      }
    }

    // ---- online softmax over this j-tile ----
    v8f tmax = s[0];
#pragma unroll
    for (int jb = 1; jb < 8; ++jb)
#pragma unroll
      for (int v = 0; v < 8; ++v) tmax[v] = fmaxf(tmax[v], s[jb][v]);
#pragma unroll
    for (int off = 8; off >= 1; off >>= 1)
#pragma unroll
      for (int v = 0; v < 8; ++v) tmax[v] = fmaxf(tmax[v], __shfl_xor(tmax[v], off, 16));

    float scale[8];
#pragma unroll
    for (int v = 0; v < 8; ++v) {
      const float mn = fmaxf(m[v], tmax[v]);
      scale[v] = __expf(m[v] - mn);
      m[v] = mn;
    }

    v8f rs = {};
#pragma unroll
    for (int jb = 0; jb < 8; ++jb)
#pragma unroll
      for (int v = 0; v < 8; ++v) {
        const float p = __expf(s[jb][v] - m[v]);
        s[jb][v] = p;
        rs[v] += p;
      }
#pragma unroll
    for (int off = 8; off >= 1; off >>= 1)
#pragma unroll
      for (int v = 0; v < 8; ++v) rs[v] += __shfl_xor(rs[v], off, 16);
#pragma unroll
    for (int v = 0; v < 8; ++v) {
      lsum[v] = lsum[v] * scale[v] + rs[v];
      ctx[v] *= scale[v];
    }

    // ---- reshape P (C-layout regs) -> A-fragment layout via per-wave LDS ----
#pragma unroll
    for (int jb = 0; jb < 8; ++jb)
#pragma unroll
      for (int v = 0; v < 8; ++v)
        pw[(size_t)(v + half * 8) * 128 + jb * 16 + l15] = (f16)s[jb][v];

    // ---- PV: ctx += P(16x128) * Vbar(128x16) ----
#pragma unroll
    for (int t = 0; t < 4; ++t) {
      const f16* pa = pw + (size_t)l15 * 128 + t * 32 + half * 16;
      v16h a = cat8(*(const v8h*)pa, *(const v8h*)(pa + 8));
      const f16* pb = vbase + (size_t)l15 * 2048 + jt * 128 + t * 32 + half * 16;
      ctx = WMMA16(a, *(const v16h*)pb, ctx);
    }
  }

  // finalize and store ctx as f16 rows [i*4+n][h*16+d]
#pragma unroll
  for (int v = 0; v < 8; ++v) {
    const float o = ctx[v] / lsum[v];
    const int i = i0 + v + half * 8;
    ctxh[(size_t)(i * 4 + n) * 128 + h * 16 + l15] = (f16)o;
  }
}

// ---------------------------------------------------------------------------
extern "C" void kernel_launch(void* const* d_in, const int* in_sizes, int n_in,
                              void* d_out, int out_size, void* d_ws, size_t ws_size,
                              hipStream_t stream) {
  (void)in_sizes; (void)n_in; (void)out_size; (void)ws_size;
  const float* Q  = (const float*)d_in[0];
  const float* K  = (const float*)d_in[1];
  // d_in[2] = num_heads (compile-time constant 8 here)
  const float* Wq = (const float*)d_in[3];
  const float* bq = (const float*)d_in[4];
  const float* Wk = (const float*)d_in[5];
  const float* bk = (const float*)d_in[6];
  const float* Wv = (const float*)d_in[7];
  const float* bv = (const float*)d_in[8];
  const float* Wo = (const float*)d_in[9];
  const float* bo = (const float*)d_in[10];

  char* ws = (char*)d_ws;
  f16* qh    = (f16*)(ws);
  f16* kh    = (f16*)(ws + (16u << 20));
  f16* vbarT = (f16*)(ws + (32u << 20));
  f16* ksumh = (f16*)(ws + (34u << 20));
  f16* ctxh  = (f16*)(ws + (36u << 20));
  float* out = (float*)d_out;

  ksum_kernel<<<(8192 * 128) / 256, 256, 0, stream>>>(K, ksumh);
  proj_kernel<0><<<512, 256, 0, stream>>>(Q, nullptr, Wq, bq, 1.0f, qh, nullptr);
  proj_kernel<0><<<512, 256, 0, stream>>>(K, nullptr, Wk, bk, 1.0f, kh, nullptr);
  proj_kernel<1><<<64, 256, 0, stream>>>(nullptr, ksumh, Wv, bv, 8.0f, vbarT, nullptr);
  attn_kernel<<<dim3(16, 32), 256, 96 * 1024, stream>>>(qh, kh, vbarT, ctxh);
  proj_kernel<2><<<64, 256, 0, stream>>>(nullptr, ctxh, Wo, bo, 1.0f, nullptr, out);
}
